// TimeResampler_2903397892391
// MI455X (gfx1250) — matrix operand order repaired and checked by reading, not compiled, not run on hardware
//
#include <hip/hip_runtime.h>
#include <cstdint>
#include <cstddef>

// ---------------- problem constants ----------------
#define B_     32
#define S_     1024
#define D_     1024
#define DEPTH_ 8
#define HEADS_ 16
#define DH_    64
#define NQ_    8
#define EMB_   768
#define FF_    4096
#define TIN_   320
#define L_     1032   // S + NQ

// ---------------- CDNA5 WMMA types ----------------
typedef __bf16 v16bf __attribute__((ext_vector_type(16)));
typedef float  v8f   __attribute__((ext_vector_type(8)));

union Frag16 {
  v16bf v;
  unsigned int u32[8];
  unsigned short u16[16];
};

__device__ __forceinline__ unsigned short f2bf(float f) {
  unsigned int u = __float_as_uint(f);
  unsigned int r = 0x7FFFu + ((u >> 16) & 1u);   // round-to-nearest-even
  return (unsigned short)((u + r) >> 16);
}
__device__ __forceinline__ float bf2f(unsigned short h) {
  return __uint_as_float(((unsigned int)h) << 16);
}

// ---------------- epilogue modes ----------------
#define EP_NONE   0
#define EP_TEMB   1   // += extra[(row/rowsPerB)*N + n]
#define EP_GELU   2   // exact gelu
#define EP_RESADD 3   // += extra[row*N + n]

// =====================================================================
// Main bf16 WMMA GEMM:  C = A(MxK,bf16,row-major) * Bt(NxK,bf16)^T
// Block tile 128x128, 8 waves (2x4), wave tile 64x32 (4x2 WMMA frags).
// Double-buffered LDS (one barrier per K-step); both operand tiles are
// K-contiguous in LDS so every fragment is two ds_load_b128 (no packing).
// Output row remap: idx = (r/rowsPerB)*batchStride + (r%rowsPerB)*ldr + baseOff + n
// Requires: M%128==0, N%128==0, K%32==0 (true for all uses here).
// =====================================================================
__global__ __launch_bounds__(256)
void k_wmma_gemm(const unsigned short* __restrict__ A,
                 const unsigned short* __restrict__ Bt,
                 float* __restrict__ Cf, unsigned short* __restrict__ Cbf,
                 const float* __restrict__ bias,
                 const float* __restrict__ extra,
                 int M, int N, int K,
                 int mode, int rowsPerB, long long batchStride, int ldr,
                 long long baseOff)
{
  __shared__ alignas(16) unsigned short As[2][128 * 32];  // [m][k]
  __shared__ alignas(16) unsigned short Bs[2][128 * 32];  // [n][k]

  const int tid  = threadIdx.x;
  const int lane = tid & 31;
  const int wid  = tid >> 5;
  const int m0 = blockIdx.y * 128;
  const int n0 = blockIdx.x * 128;
  const int mwOff = (wid >> 2) * 64;  // 0 / 64
  const int nwOff = (wid & 3) * 32;   // 0 / 32 / 64 / 96

  v8f zero = {};
  v8f acc[4][2];
#pragma unroll
  for (int i = 0; i < 4; i++)
#pragma unroll
    for (int j = 0; j < 2; j++) acc[i][j] = zero;

  // cooperative-load mapping: each thread moves 16 bf16 per tile per operand
  const int ldRow = tid >> 1;          // 0..127
  const int ldCol = (tid & 1) * 16;    // 0 / 16

  const unsigned short* Aptr = A  + (size_t)(m0 + ldRow) * K + ldCol;
  const unsigned short* Bptr = Bt + (size_t)(n0 + ldRow) * K + ldCol;

  // ISA bf16 A layout: lane l<16 row=l, K chunks {kb..kb+3, kb+4..7, kb+16.., kb+20..}
  const int rowBase = mwOff + (lane & 15);
  const int kbA = (lane & 16) ? 8 : 0;
  const int q0  = kbA >> 2;                 // uint2 index of first A chunk
  // bf16 B layout: lanes 0-15 own K 0..7,16..23; lanes 16-31 own K 8..15,24..31 (matches A)
  const int kbB = (lane & 16) ? 8 : 0;

  uint4 av0, av1, bv0, bv1;
  // ---- preload tile 0 ----
  {
    const uint4* ag = (const uint4*)(Aptr);
    av0 = ag[0]; av1 = ag[1];
    const uint4* bg = (const uint4*)(Bptr);
    bv0 = bg[0]; bv1 = bg[1];
  }
  *(uint4*)(&As[0][ldRow * 32 + ldCol])     = av0;
  *(uint4*)(&As[0][ldRow * 32 + ldCol + 8]) = av1;
  *(uint4*)(&Bs[0][ldRow * 32 + ldCol])     = bv0;
  *(uint4*)(&Bs[0][ldRow * 32 + ldCol + 8]) = bv1;
  __syncthreads();

  int p = 0;
  for (int k0 = 0; k0 < K; k0 += 32) {
    const bool more = (k0 + 32 < K);
    if (more) {
      const uint4* ag = (const uint4*)(Aptr + (k0 + 32));
      av0 = ag[0]; av1 = ag[1];
      const uint4* bg = (const uint4*)(Bptr + (k0 + 32));
      bv0 = bg[0]; bv1 = bg[1];
      if (k0 + 64 < K) {  // gfx1250 global_prefetch_b8, two tiles ahead
        __builtin_prefetch(Aptr + (k0 + 64), 0, 3);
        __builtin_prefetch(Bptr + (k0 + 64), 0, 3);
      }
    }

    // ---- fragments (all ds_load_b128-shaped) ----
    const unsigned short* Ap = As[p];
    const unsigned short* Bp = Bs[p];
    Frag16 a[4], bfr[2];
#pragma unroll
    for (int mf = 0; mf < 4; mf++) {
      const uint2* ap = (const uint2*)(Ap + (rowBase + mf * 16) * 32);
      uint2 t0 = ap[q0];
      uint2 t1 = ap[q0 + 1];
      uint2 t2 = ap[q0 + 4];
      uint2 t3 = ap[q0 + 5];
      a[mf].u32[0] = t0.x; a[mf].u32[1] = t0.y;
      a[mf].u32[2] = t1.x; a[mf].u32[3] = t1.y;
      a[mf].u32[4] = t2.x; a[mf].u32[5] = t2.y;
      a[mf].u32[6] = t3.x; a[mf].u32[7] = t3.y;
    }
#pragma unroll
    for (int nf = 0; nf < 2; nf++) {
      const int col = nwOff + nf * 16 + (lane & 15);
      const uint4* bp = (const uint4*)(Bp + col * 32 + kbB);
      uint4 u0 = bp[0];
      uint4 u1 = bp[2];
      bfr[nf].u32[0] = u0.x; bfr[nf].u32[1] = u0.y;
      bfr[nf].u32[2] = u0.z; bfr[nf].u32[3] = u0.w;
      bfr[nf].u32[4] = u1.x; bfr[nf].u32[5] = u1.y;
      bfr[nf].u32[6] = u1.z; bfr[nf].u32[7] = u1.w;
    }

#pragma unroll
    for (int nf = 0; nf < 2; nf++)
#pragma unroll
      for (int mf = 0; mf < 4; mf++)
        acc[mf][nf] = __builtin_amdgcn_wmma_f32_16x16x32_bf16(
            false, a[mf].v, false, bfr[nf].v, (short)0, acc[mf][nf], false, false);

    if (more) {
      const int pn = p ^ 1;
      *(uint4*)(&As[pn][ldRow * 32 + ldCol])     = av0;
      *(uint4*)(&As[pn][ldRow * 32 + ldCol + 8]) = av1;
      *(uint4*)(&Bs[pn][ldRow * 32 + ldCol])     = bv0;
      *(uint4*)(&Bs[pn][ldRow * 32 + ldCol + 8]) = bv1;
    }
    __syncthreads();
    p ^= 1;
  }

  // C/D layout: lanes 0-15 N=lane M=r; lanes 16-31 N=lane-16 M=r+8
  const int colL   = lane & 15;
  const int hiHalf = (lane >> 4) & 1;
#pragma unroll
  for (int mf = 0; mf < 4; mf++)
#pragma unroll
    for (int nf = 0; nf < 2; nf++)
#pragma unroll
      for (int r = 0; r < 8; r++) {
        const int gm = m0 + mwOff + mf * 16 + r + hiHalf * 8;
        const int gn = n0 + nwOff + nf * 16 + colL;
        float v = acc[mf][nf][r];
        if (bias) v += bias[gn];
        if (mode == EP_TEMB)        v += extra[(size_t)(gm / rowsPerB) * N + gn];
        else if (mode == EP_RESADD) v += extra[(size_t)gm * N + gn];
        else if (mode == EP_GELU)   v = 0.5f * v * (1.0f + erff(v * 0.70710678118654752f));
        const long long idx = (long long)(gm / rowsPerB) * batchStride +
                              (long long)(gm % rowsPerB) * ldr + baseOff + gn;
        if (Cf)  Cf[idx]  = v;
        if (Cbf) Cbf[idx] = f2bf(v);
      }
}

// =====================================================================
// Naive fp32 GEMM for small ops. act: 0 none, 1 silu. Same row remap.
// =====================================================================
__global__ void k_gemm_naive(const float* __restrict__ A, const float* __restrict__ Bm,
                             const float* __restrict__ bias,
                             float* __restrict__ Cf, unsigned short* __restrict__ Cbf,
                             int M, int N, int K, int act,
                             int rowsPerB, long long batchStride, int ldr,
                             long long baseOff)
{
  const int n = blockIdx.x * blockDim.x + threadIdx.x;
  const int r = blockIdx.y;
  if (n >= N || r >= M) return;
  const float* ar = A + (size_t)r * K;
  float s = bias ? bias[n] : 0.0f;
  for (int k = 0; k < K; k++) s += ar[k] * Bm[(size_t)k * N + n];
  if (act == 1) s = s / (1.0f + __expf(-s));  // silu
  const long long idx = (long long)(r / rowsPerB) * batchStride +
                        (long long)(r % rowsPerB) * ldr + baseOff + n;
  if (Cf)  Cf[idx]  = s;
  if (Cbf) Cbf[idx] = f2bf(s);
}

// =====================================================================
// LayerNorm (+ optional affine, + optional adaLN modulation), fp32/bf16 out
// one block (256 threads) per row of D=1024
// =====================================================================
__global__ __launch_bounds__(256)
void k_ln_mod(const float* __restrict__ x, int D,
              const float* __restrict__ w, const float* __restrict__ b,
              const float* __restrict__ sh, const float* __restrict__ sc,
              int adaStride, int rowsPerBatch,
              float* __restrict__ outF, unsigned short* __restrict__ outBF)
{
  const int row = blockIdx.x;
  const int tid = threadIdx.x;
  const float* xr = x + (size_t)row * D;
  __shared__ float red[256];

  float s = 0.0f;
  for (int d = tid; d < D; d += 256) s += xr[d];
  red[tid] = s; __syncthreads();
  for (int off = 128; off > 0; off >>= 1) {
    if (tid < off) red[tid] += red[tid + off];
    __syncthreads();
  }
  const float mean = red[0] / (float)D;
  __syncthreads();

  float vs = 0.0f;
  for (int d = tid; d < D; d += 256) { float t = xr[d] - mean; vs += t * t; }
  red[tid] = vs; __syncthreads();
  for (int off = 128; off > 0; off >>= 1) {
    if (tid < off) red[tid] += red[tid + off];
    __syncthreads();
  }
  const float inv = rsqrtf(red[0] / (float)D + 1e-5f);

  const int bb = row / rowsPerBatch;
  for (int d = tid; d < D; d += 256) {
    float y = (xr[d] - mean) * inv;
    if (w)  y = y * w[d] + b[d];
    if (sc) y = y * (1.0f + sc[(size_t)bb * adaStride + d]) + sh[(size_t)bb * adaStride + d];
    if (outF)  outF[(size_t)row * D + d] = y;
    if (outBF) outBF[(size_t)row * D + d] = f2bf(y);
  }
}

// =====================================================================
// Attention: one block per (b, head, query); 1032 keys, fp32 softmax,
// bf16 K/V read from the packed kv buffer kv[b][t][0:2048] (k | v halves)
// =====================================================================
__global__ __launch_bounds__(256)
void k_attn(const float* __restrict__ q, const unsigned short* __restrict__ kv,
            float* __restrict__ o)
{
  const int idx = blockIdx.x;
  const int b  = idx >> 7;        // /(HEADS*NQ)
  const int h  = (idx >> 3) & 15;
  const int qi = idx & 7;
  const int tid = threadIdx.x;

  __shared__ float qs[DH_];
  __shared__ float scs[L_];
  __shared__ float red[256];
  __shared__ float part[4][DH_];

  const float* qp = q + ((size_t)b * NQ_ + qi) * D_ + h * DH_;
  if (tid < DH_) qs[tid] = qp[tid];
  __syncthreads();

  const unsigned short* kvb = kv + (size_t)b * L_ * 2048;
  for (int t = tid; t < L_; t += 256) {
    const unsigned short* kp = kvb + (size_t)t * 2048 + h * DH_;
    float s = 0.0f;
#pragma unroll
    for (int d = 0; d < DH_; d++) s += qs[d] * bf2f(kp[d]);
    scs[t] = s * 0.125f;  // scale^2 = DH^-0.5
  }
  __syncthreads();

  float m = -3.4e38f;
  for (int t = tid; t < L_; t += 256) m = fmaxf(m, scs[t]);
  red[tid] = m; __syncthreads();
  for (int off = 128; off > 0; off >>= 1) {
    if (tid < off) red[tid] = fmaxf(red[tid], red[tid + off]);
    __syncthreads();
  }
  m = red[0]; __syncthreads();

  float ssum = 0.0f;
  for (int t = tid; t < L_; t += 256) {
    float e = __expf(scs[t] - m);
    scs[t] = e;
    ssum += e;
  }
  red[tid] = ssum; __syncthreads();
  for (int off = 128; off > 0; off >>= 1) {
    if (tid < off) red[tid] += red[tid + off];
    __syncthreads();
  }
  const float inv = 1.0f / red[0];
  __syncthreads();

  const int d  = tid & 63;
  const int ch = tid >> 6;  // 4 chunks over keys
  float acc = 0.0f;
  for (int t = ch; t < L_; t += 4)
    acc += scs[t] * bf2f(kvb[(size_t)t * 2048 + 1024 + h * DH_ + d]);
  part[ch][d] = acc; __syncthreads();
  if (tid < DH_) {
    float r = (part[0][tid] + part[1][tid] + part[2][tid] + part[3][tid]) * inv;
    o[((size_t)b * NQ_ + qi) * D_ + h * DH_ + tid] = r;
  }
}

// ---------------- small elementwise kernels ----------------
__global__ void k_time_proj(const int* __restrict__ t, float* __restrict__ tp)
{
  const int b = blockIdx.x, j = threadIdx.x;  // block 160
  const float freq = __expf(-logf(10000.0f) * (float)j / 160.0f);
  const float e = (float)t[b] * freq;
  tp[b * TIN_ + j]        = cosf(e);   // flip_sin_to_cos: cos first
  tp[b * TIN_ + 160 + j]  = sinf(e);
}

__global__ void k_silu(const float* __restrict__ x, float* __restrict__ y, int n)
{
  int i = blockIdx.x * blockDim.x + threadIdx.x;
  if (i < n) { float v = x[i]; y[i] = v / (1.0f + __expf(-v)); }
}

__global__ void k_f2bf(const float* __restrict__ x, unsigned short* __restrict__ y, long long n)
{
  long long i = (long long)blockIdx.x * blockDim.x + threadIdx.x;
  if (i < n) y[i] = f2bf(x[i]);
}

// Tiled transpose + bf16 convert (+ optional per-K row scale, for the ln1
// fold): y[n][k] = bf16(rowScale[k] * w[k][n]).  32x32 tiles, block (32,8).
__global__ void k_f2bf_t(const float* __restrict__ w, unsigned short* __restrict__ y,
                         int Kd, int Nd, const float* __restrict__ rowScale)
{
  __shared__ unsigned short tile[32][33];
  const int n0 = blockIdx.x * 32;
  const int k0 = blockIdx.y * 32;
  const int tx = threadIdx.x, ty = threadIdx.y;
#pragma unroll
  for (int i = 0; i < 32; i += 8) {
    const int k = k0 + ty + i;
    float v = w[(size_t)k * Nd + (n0 + tx)];
    if (rowScale) v *= rowScale[k];
    tile[ty + i][tx] = f2bf(v);
  }
  __syncthreads();
#pragma unroll
  for (int i = 0; i < 32; i += 8) {
    y[(size_t)(n0 + ty + i) * Kd + (k0 + tx)] = tile[tx][ty + i];
  }
}

__global__ void k_bcast_lat(const float* __restrict__ lat0, float* __restrict__ lat)
{
  long long i = (long long)blockIdx.x * blockDim.x + threadIdx.x;
  if (i < (long long)B_ * NQ_ * D_) lat[i] = lat0[i % (NQ_ * D_)];
}

// =====================================================================
extern "C" void kernel_launch(void* const* d_in, const int* in_sizes, int n_in,
                              void* d_out, int out_size, void* d_ws, size_t ws_size,
                              hipStream_t stream)
{
  (void)in_sizes; (void)n_in; (void)out_size; (void)ws_size;

  const float* x          = (const float*)d_in[0];
  const int*   timestep   = (const int*)  d_in[1];
  const float* latents0   = (const float*)d_in[2];
  const float* proj_in_w  = (const float*)d_in[3];
  const float* proj_in_b  = (const float*)d_in[4];
  const float* t1_w       = (const float*)d_in[5];
  const float* t1_b       = (const float*)d_in[6];
  const float* t2_w       = (const float*)d_in[7];
  const float* t2_b       = (const float*)d_in[8];
  const float* ln1_w      = (const float*)d_in[9];
  const float* ln1_b      = (const float*)d_in[10];
  const float* ln2_w      = (const float*)d_in[11];
  const float* ln2_b      = (const float*)d_in[12];
  const float* wq         = (const float*)d_in[13];
  const float* wkv        = (const float*)d_in[14];
  const float* wout       = (const float*)d_in[15];
  const float* ffln_w     = (const float*)d_in[16];
  const float* ffln_b     = (const float*)d_in[17];
  const float* ff_w1      = (const float*)d_in[18];
  const float* ff_w2      = (const float*)d_in[19];
  const float* ada_w      = (const float*)d_in[20];
  const float* ada_b      = (const float*)d_in[21];
  const float* proj_out_w = (const float*)d_in[22];
  const float* proj_out_b = (const float*)d_in[23];
  const float* norm_out_w = (const float*)d_in[24];
  const float* norm_out_b = (const float*)d_in[25];

  // ---- workspace carve-out ----
  char* ws = (char*)d_ws;
  size_t off = 0;
  auto alloc = [&](size_t bytes) -> char* {
    char* p = ws + off;
    off = (off + bytes + 255) & ~(size_t)255;
    return p;
  };

  float* tp    = (float*)alloc((size_t)B_ * TIN_ * 4);
  float* tmp1  = (float*)alloc((size_t)B_ * D_ * 4);
  float* temb  = (float*)alloc((size_t)B_ * D_ * 4);
  float* stemb = (float*)alloc((size_t)B_ * D_ * 4);
  float* ada   = (float*)alloc((size_t)B_ * 4 * D_ * 4);
  float* fb    = (float*)alloc((size_t)2 * D_ * 4);
  float* lat   = (float*)alloc((size_t)B_ * NQ_ * D_ * 4);
  float* latmod = (float*)alloc((size_t)B_ * NQ_ * D_ * 4);
  unsigned short* latmod_bf = (unsigned short*)alloc((size_t)B_ * NQ_ * D_ * 2);
  float* qbuf  = (float*)alloc((size_t)B_ * NQ_ * D_ * 4);
  float* obuf  = (float*)alloc((size_t)B_ * NQ_ * D_ * 4);
  unsigned short* obf = (unsigned short*)alloc((size_t)B_ * NQ_ * D_ * 2);
  unsigned short* fbf = (unsigned short*)alloc((size_t)B_ * NQ_ * D_ * 2);
  float* gbuf  = (float*)alloc((size_t)B_ * NQ_ * FF_ * 4);
  unsigned short* gbf = (unsigned short*)alloc((size_t)B_ * NQ_ * FF_ * 2);
  float* ybuf  = (float*)alloc((size_t)B_ * NQ_ * D_ * 4);
  unsigned short* wbuf = (unsigned short*)alloc((size_t)D_ * FF_ * 2);  // transposed weight staging (max 4M elem)
  unsigned short* xbf  = (unsigned short*)alloc((size_t)B_ * S_ * EMB_ * 2);
  unsigned short* hnorm = (unsigned short*)alloc((size_t)B_ * S_ * D_ * 2);
  // union: h(fp32, pre-loop) overlaps kv(bf16, in-loop) — h is dead once hnorm exists
  size_t hBytes  = (size_t)B_ * S_ * D_ * 4;
  size_t kvBytes = (size_t)B_ * L_ * 2048 * 2;
  char* uni = alloc(hBytes > kvBytes ? hBytes : kvBytes);
  float* h = (float*)uni;
  unsigned short* kvb = (unsigned short*)uni;

  const int BIGROW = 1 << 30;  // "no batching" row remap
  dim3 tb32x8(32, 8);

  // ---- time embedding: Timesteps -> Linear -> SiLU -> Linear ----
  k_time_proj<<<B_, 160, 0, stream>>>(timestep, tp);
  dim3 gn1((D_ + 255) / 256, B_);
  k_gemm_naive<<<gn1, 256, 0, stream>>>(tp, t1_w, t1_b, tmp1, nullptr,
      B_, D_, TIN_, 1, BIGROW, 0, D_, 0);
  k_gemm_naive<<<gn1, 256, 0, stream>>>(tmp1, t2_w, t2_b, temb, nullptr,
      B_, D_, D_, 0, BIGROW, 0, D_, 0);
  k_silu<<<(B_ * D_ + 255) / 256, 256, 0, stream>>>(temb, stemb, B_ * D_);

  // ---- proj_in: h = x @ W + b + temb[b]  (bf16 WMMA, fp32 accum) ----
  const long long nx = (long long)B_ * S_ * EMB_;
  k_f2bf<<<(unsigned)((nx + 255) / 256), 256, 0, stream>>>(x, xbf, nx);
  k_f2bf_t<<<dim3(D_ / 32, EMB_ / 32), tb32x8, 0, stream>>>(proj_in_w, wbuf, EMB_, D_, nullptr);
  dim3 gPin(D_ / 128, (B_ * S_) / 128);
  k_wmma_gemm<<<gPin, 256, 0, stream>>>(xbf, wbuf, h, nullptr, proj_in_b, temb,
      B_ * S_, D_, EMB_, EP_TEMB, S_, (long long)S_ * D_, D_, 0);

  // ---- hnorm = normalize(h) (no affine; ln1 affine folded into wkv) ----
  k_ln_mod<<<B_ * S_, 256, 0, stream>>>(h, D_, nullptr, nullptr, nullptr, nullptr,
      0, 1, nullptr, hnorm);

  // ---- lat = broadcast(latents0) ----
  k_bcast_lat<<<(B_ * NQ_ * D_ + 255) / 256, 256, 0, stream>>>(latents0, lat);

  for (int l = 0; l < DEPTH_; l++) {
    const float* wq_l   = wq   + (size_t)l * D_ * D_;
    const float* wkv_l  = wkv  + (size_t)l * D_ * 2 * D_;
    const float* wout_l = wout + (size_t)l * D_ * D_;
    const float* ffw1_l = ff_w1 + (size_t)l * D_ * FF_;
    const float* ffw2_l = ff_w2 + (size_t)l * FF_ * D_;
    const float* adaw_l = ada_w + (size_t)l * D_ * 4 * D_;
    const float* adab_l = ada_b + (size_t)l * 4 * D_;
    const float* ln1w_l = ln1_w + l * D_;
    const float* ln1b_l = ln1_b + l * D_;
    const float* ln2w_l = ln2_w + l * D_;
    const float* ln2b_l = ln2_b + l * D_;
    const float* fflnw_l = ffln_w + l * D_;
    const float* fflnb_l = ffln_b + l * D_;

    // adaLN: ada = silu(temb) @ ada_w + ada_b  -> [sh_msa|sc_msa|sh_mlp|sc_mlp]
    dim3 gna((4 * D_ + 255) / 256, B_);
    k_gemm_naive<<<gna, 256, 0, stream>>>(stemb, adaw_l, adab_l, ada, nullptr,
        B_, 4 * D_, D_, 0, BIGROW, 0, 4 * D_, 0);

    // fold ln1 affine into wkv: wbuf = bf16(diag(ln1_w)@wkv)^T ; fb = ln1_b @ wkv
    k_f2bf_t<<<dim3((2 * D_) / 32, D_ / 32), tb32x8, 0, stream>>>(wkv_l, wbuf, D_, 2 * D_, ln1w_l);
    dim3 gnf((2 * D_ + 255) / 256, 1);
    k_gemm_naive<<<gnf, 256, 0, stream>>>(ln1b_l, wkv_l, nullptr, fb, nullptr,
        1, 2 * D_, D_, 0, BIGROW, 0, 2 * D_, 0);

    // KV over sequence (the 138 GFLOP GEMM): kv[b][0..1023][:] = hnorm @ wkv' + fb
    dim3 gkv((2 * D_) / 128, (B_ * S_) / 128);
    k_wmma_gemm<<<gkv, 256, 0, stream>>>(hnorm, wbuf, nullptr, kvb, fb, nullptr,
        B_ * S_, 2 * D_, D_, EP_NONE, S_, (long long)L_ * 2048, 2048, 0);

    // lat_mod = LN(lat; ln2) * (1 + sc_msa) + sh_msa
    k_ln_mod<<<B_ * NQ_, 256, 0, stream>>>(lat, D_, ln2w_l, ln2b_l,
        ada /*sh_msa*/, ada + D_ /*sc_msa*/, 4 * D_, NQ_, latmod, latmod_bf);

    // kv[b][1024..1031][:] = lat_mod @ wkv (unfolded; lat path has its own LN)
    dim3 gnk((2 * D_ + 255) / 256, B_ * NQ_);
    k_gemm_naive<<<gnk, 256, 0, stream>>>(latmod, wkv_l, nullptr, nullptr, kvb,
        B_ * NQ_, 2 * D_, D_, 0, NQ_, (long long)L_ * 2048, 2048, (long long)1024 * 2048);

    // q = lat_mod @ wq
    k_f2bf_t<<<dim3(D_ / 32, D_ / 32), tb32x8, 0, stream>>>(wq_l, wbuf, D_, D_, nullptr);
    dim3 gq(D_ / 128, (B_ * NQ_) / 128);
    k_wmma_gemm<<<gq, 256, 0, stream>>>(latmod_bf, wbuf, qbuf, nullptr, nullptr, nullptr,
        B_ * NQ_, D_, D_, EP_NONE, BIGROW, 0, D_, 0);

    // attention
    k_attn<<<B_ * HEADS_ * NQ_, 256, 0, stream>>>(qbuf, kvb, obuf);

    // lat = lat + o @ wout
    k_f2bf<<<(B_ * NQ_ * D_ + 255) / 256, 256, 0, stream>>>(obuf, obf, (long long)B_ * NQ_ * D_);
    k_f2bf_t<<<dim3(D_ / 32, D_ / 32), tb32x8, 0, stream>>>(wout_l, wbuf, D_, D_, nullptr);
    k_wmma_gemm<<<gq, 256, 0, stream>>>(obf, wbuf, lat, nullptr, nullptr, lat,
        B_ * NQ_, D_, D_, EP_RESADD, BIGROW, 0, D_, 0);

    // FF: f = LN(lat; ffln)*(1+sc_mlp)+sh_mlp ; lat += gelu(f@W1)@W2
    k_ln_mod<<<B_ * NQ_, 256, 0, stream>>>(lat, D_, fflnw_l, fflnb_l,
        ada + 2 * D_ /*sh_mlp*/, ada + 3 * D_ /*sc_mlp*/, 4 * D_, NQ_, nullptr, fbf);

    k_f2bf_t<<<dim3(FF_ / 32, D_ / 32), tb32x8, 0, stream>>>(ffw1_l, wbuf, D_, FF_, nullptr);
    dim3 gf1(FF_ / 128, (B_ * NQ_) / 128);
    k_wmma_gemm<<<gf1, 256, 0, stream>>>(fbf, wbuf, gbuf, nullptr, nullptr, nullptr,
        B_ * NQ_, FF_, D_, EP_GELU, BIGROW, 0, FF_, 0);

    k_f2bf<<<(unsigned)(((long long)B_ * NQ_ * FF_ + 255) / 256), 256, 0, stream>>>(
        gbuf, gbf, (long long)B_ * NQ_ * FF_);
    k_f2bf_t<<<dim3(D_ / 32, FF_ / 32), tb32x8, 0, stream>>>(ffw2_l, wbuf, FF_, D_, nullptr);
    k_wmma_gemm<<<gq, 256, 0, stream>>>(gbf, wbuf, lat, nullptr, nullptr, lat,
        B_ * NQ_, D_, FF_, EP_RESADD, BIGROW, 0, D_, 0);
  }

  // ---- out = LN(lat @ proj_out_w + proj_out_b; norm_out) ----
  dim3 gno((D_ + 255) / 256, B_ * NQ_);
  k_gemm_naive<<<gno, 256, 0, stream>>>(lat, proj_out_w, proj_out_b, ybuf, nullptr,
      B_ * NQ_, D_, D_, 0, BIGROW, 0, D_, 0);
  k_ln_mod<<<B_ * NQ_, 256, 0, stream>>>(ybuf, D_, norm_out_w, norm_out_b,
      nullptr, nullptr, 0, 1, (float*)d_out, nullptr);
}
